// FF_NLP_WTA_15324443312628
// MI455X (gfx1250) — compile-verified, matrix operand order checked by reading, not run on hardware
//
#include <hip/hip_runtime.h>
#include <hip/hip_bf16.h>

// Problem constants (T=4096, B=8 -> M = 32768 rows)
#define M_ROWS 32768
#define IN_I   256
#define HID    512
#define CONC   1024
#define OUT_O  512
#define HPAD   520   // 512 + 8 bf16 pad -> 4-bank row skew, conflict-free b128 LDS reads
#define W1PAD  264   // 256 + 8 bf16 pad (row stride 528 B)

typedef __attribute__((ext_vector_type(16))) __bf16 v16bf;
typedef __attribute__((ext_vector_type(8)))  __bf16 v8bf;
typedef __attribute__((ext_vector_type(8)))  float  v8f;

__device__ __forceinline__ v8f wmma_bf16(v16bf a, v16bf b, v8f c) {
  // D = A(16x32 bf16) * B(32x16 bf16) + C(16x16 f32)
  return __builtin_amdgcn_wmma_f32_16x16x32_bf16(false, a, false, b, (short)0, c,
                                                 false, false);
}

// ---- CDNA5 async global->LDS copy (ASYNCcnt-tracked), ISA cdna5 ch.10/15.18 ----
__device__ __forceinline__ void async_b128(unsigned lds_addr, const void* gptr) {
  unsigned long long ga = (unsigned long long)(uintptr_t)gptr;
  asm volatile("global_load_async_to_lds_b128 %0, %1, off"
               :: "v"(lds_addr), "v"(ga) : "memory");
}
__device__ __forceinline__ void wait_async0() {
  asm volatile("s_wait_asynccnt 0x0" ::: "memory");
}

// A/B fragment (ISA 7.12.2, 16-bit 16x32): lane L<16 holds row L, K={k0..k0+7,k0+16..k0+23};
// lane L>=16 holds row L-16 with K shifted by 8 (caller folds the shift into k0).
__device__ __forceinline__ v16bf frag_from_f32(const float* p, int k0) {
  const float4 a = *reinterpret_cast<const float4*>(p + k0);
  const float4 b = *reinterpret_cast<const float4*>(p + k0 + 4);
  const float4 c = *reinterpret_cast<const float4*>(p + k0 + 16);
  const float4 d = *reinterpret_cast<const float4*>(p + k0 + 20);
  v16bf f;
  f[0]=(__bf16)a.x;  f[1]=(__bf16)a.y;  f[2]=(__bf16)a.z;  f[3]=(__bf16)a.w;
  f[4]=(__bf16)b.x;  f[5]=(__bf16)b.y;  f[6]=(__bf16)b.z;  f[7]=(__bf16)b.w;
  f[8]=(__bf16)c.x;  f[9]=(__bf16)c.y;  f[10]=(__bf16)c.z; f[11]=(__bf16)c.w;
  f[12]=(__bf16)d.x; f[13]=(__bf16)d.y; f[14]=(__bf16)d.z; f[15]=(__bf16)d.w;
  return f;
}

__device__ __forceinline__ v16bf frag_from_bf16(const __bf16* p, int k0) {
  v8bf lo = *reinterpret_cast<const v8bf*>(p + k0);        // 16B (ds/global b128)
  v8bf hi = *reinterpret_cast<const v8bf*>(p + k0 + 16);   // 16B
  return __builtin_shufflevector(lo, hi, 0,1,2,3,4,5,6,7,8,9,10,11,12,13,14,15);
}

// ---------------- weight fp32 -> bf16 conversion ----------------
__global__ __launch_bounds__(256) void cvt_kernel(const float* __restrict__ s,
                                                  __bf16* __restrict__ d, int n) {
  int i = blockIdx.x * 256 + threadIdx.x;
  if (i < n) d[i] = (__bf16)s[i];
}

// -------- per-concept logsumexp table: L[c] = LSE_o(h2o_w[o,c]+h2o_b[o]) --------
__global__ __launch_bounds__(256) void lse_kernel(const float* __restrict__ w3,
                                                  const float* __restrict__ b3,
                                                  float* __restrict__ L) {
  __shared__ float red[256];
  const int c = blockIdx.x;
  const int t = threadIdx.x;
  float mx = -3.4e38f;
  for (int o = t; o < OUT_O; o += 256)
    mx = fmaxf(mx, w3[(size_t)o * CONC + c] + b3[o]);
  red[t] = mx;
  __syncthreads();
  for (int s = 128; s > 0; s >>= 1) {
    if (t < s) red[t] = fmaxf(red[t], red[t + s]);
    __syncthreads();
  }
  mx = red[0];
  __syncthreads();
  float sum = 0.f;
  for (int o = t; o < OUT_O; o += 256)
    sum += __expf(w3[(size_t)o * CONC + c] + b3[o] - mx);
  red[t] = sum;
  __syncthreads();
  for (int s = 128; s > 0; s >>= 1) {
    if (t < s) red[t] += red[t + s];
    __syncthreads();
  }
  if (t == 0) L[c] = mx + __logf(red[0]);
}

// ---- fused GEMM1(relu) -> LDS -> GEMM2(relu) -> argmax, async-staged weights ----
// 4 waves/block, each wave owns 16 rows. Weight tiles double-buffered in LDS via
// global_load_async_to_lds_b128; all WMMA under uniform EXEC.
__global__ __launch_bounds__(128) void ff_wta_kernel(const float* __restrict__ x,
                                                     const __bf16* __restrict__ w1b,
                                                     const float* __restrict__ b1,
                                                     const __bf16* __restrict__ w2b,
                                                     const float* __restrict__ b2,
                                                     int* __restrict__ aidx) {
  extern __shared__ char smem[];
  const unsigned lds0 = (unsigned)(uintptr_t)(void*)smem;  // LDS byte offset (addr[31:0])
  const int t    = threadIdx.x;
  const int lane = t & 31;
  const int wave = t >> 5;
  const int lrow = lane & 15;      // M (A) or N (B/D) index
  const int half = lane >> 4;      // K-half selector
  const int koff = half * 8;
  const int m0 = (blockIdx.x * 4 + wave) * 16;

  // LDS layout (bytes): h[4][16][HPAD]  |  w1 stage x2  |  w2 stage x2
  const unsigned W1S = 4u * 16u * HPAD * 2u;          // 66560
  const unsigned W2S = W1S + 2u * 16u * W1PAD * 2u;   // +16896 = 83456
  __bf16* hl = reinterpret_cast<__bf16*>(smem) + wave * 16 * HPAD;

  // ---- GEMM1: h[16,512] = relu(x[16,256] @ W1^T + b1), bf16 into LDS ----
  const float* xrow = x + (size_t)(m0 + lrow) * IN_I;
  v16bf a1[8];
  #pragma unroll
  for (int ks = 0; ks < 8; ++ks)
    a1[ks] = frag_from_f32(xrow, ks * 32 + koff);

  {
    // stage W1 tile nb=0 into buffer 0: 16 rows x 512B, 64B per thread
    const int r  = t >> 3;
    const int co = (t & 7) * 64;
    const char* g = (const char*)(w1b + (size_t)r * IN_I) + co;
    unsigned l = lds0 + W1S + (unsigned)(r * 528 + co);
    #pragma unroll
    for (int j = 0; j < 4; ++j) async_b128(l + j * 16, g + j * 16);
  }
  wait_async0();
  __syncthreads();

  for (int nb = 0; nb < 32; ++nb) {
    if (nb + 1 < 32) {  // prefetch next tile into the other buffer
      const int r  = t >> 3;
      const int co = (t & 7) * 64;
      const char* g = (const char*)(w1b + (size_t)((nb + 1) * 16 + r) * IN_I) + co;
      unsigned l = lds0 + W1S + (unsigned)(((nb + 1) & 1) * 8448 + r * 528 + co);
      #pragma unroll
      for (int j = 0; j < 4; ++j) async_b128(l + j * 16, g + j * 16);
    }
    const __bf16* wrow =
        reinterpret_cast<const __bf16*>(smem + W1S + (nb & 1) * 8448) + lrow * W1PAD;
    const int n = nb * 16 + lrow;
    const float bv = b1[n];
    v8f acc = {bv, bv, bv, bv, bv, bv, bv, bv};
    #pragma unroll
    for (int ks = 0; ks < 8; ++ks)
      acc = wmma_bf16(a1[ks], frag_from_bf16(wrow, ks * 32 + koff), acc);
    // D layout: lane holds column n, rows (half*8 + v)
    #pragma unroll
    for (int v = 0; v < 8; ++v) {
      float r = fmaxf(acc[v], 0.f);
      hl[(half * 8 + v) * HPAD + n] = (__bf16)r;
    }
    wait_async0();
    __syncthreads();
  }

  // ---- GEMM2: c[16,1024] = relu(h @ W2^T + b2), fused running argmax ----
  const __bf16* hrow = hl + lrow * HPAD;
  v16bf a2[16];
  #pragma unroll
  for (int ks = 0; ks < 16; ++ks)
    a2[ks] = frag_from_bf16(hrow, ks * 32 + koff);

  float bestv[8];
  int   bestn[8];
  #pragma unroll
  for (int v = 0; v < 8; ++v) { bestv[v] = -1.f; bestn[v] = 0; }

  {
    // stage W2 tile nb=0 into buffer 0: 16 rows x 1024B, 128B per thread
    const int r  = t >> 3;
    const int co = (t & 7) * 128;
    const char* g = (const char*)(w2b + (size_t)r * HID) + co;
    unsigned l = lds0 + W2S + (unsigned)(r * 1040 + co);
    #pragma unroll
    for (int j = 0; j < 8; ++j) async_b128(l + j * 16, g + j * 16);
  }
  wait_async0();
  __syncthreads();

  for (int nb = 0; nb < 64; ++nb) {
    if (nb + 1 < 64) {
      const int r  = t >> 3;
      const int co = (t & 7) * 128;
      const char* g = (const char*)(w2b + (size_t)((nb + 1) * 16 + r) * HID) + co;
      unsigned l = lds0 + W2S + (unsigned)(((nb + 1) & 1) * 16640 + r * 1040 + co);
      #pragma unroll
      for (int j = 0; j < 8; ++j) async_b128(l + j * 16, g + j * 16);
    }
    const __bf16* wrow =
        reinterpret_cast<const __bf16*>(smem + W2S + (nb & 1) * 16640) + lrow * HPAD;
    const int n = nb * 16 + lrow;
    const float bv = b2[n];
    v8f acc = {bv, bv, bv, bv, bv, bv, bv, bv};
    #pragma unroll
    for (int ks = 0; ks < 16; ++ks)
      acc = wmma_bf16(a2[ks], frag_from_bf16(wrow, ks * 32 + koff), acc);
    #pragma unroll
    for (int v = 0; v < 8; ++v) {
      float r = fmaxf(acc[v], 0.f);
      if (r > bestv[v]) { bestv[v] = r; bestn[v] = n; }
    }
    wait_async0();
    __syncthreads();
  }

  // ---- cross-lane argmax: each row's 16 residue-candidates live in 16 lanes ----
  float* sv = reinterpret_cast<float*>(hl);        // reuse this wave's private LDS stripe
  int*   si = reinterpret_cast<int*>(hl) + 256;
  #pragma unroll
  for (int v = 0; v < 8; ++v) {
    const int r = half * 8 + v;
    sv[r * 16 + lrow] = bestv[v];
    si[r * 16 + lrow] = bestn[v];
  }
  __syncthreads();
  if (lane < 16) {
    float bv = sv[lane * 16];
    int   bn = si[lane * 16];
    #pragma unroll
    for (int j = 1; j < 16; ++j) {
      float cv = sv[lane * 16 + j];
      int   cn = si[lane * 16 + j];
      if (cv > bv || (cv == bv && cn < bn)) { bv = cv; bn = cn; }
    }
    aidx[m0 + lane] = bn;
  }
}

// ---------------- output: out[row,:] = h2o_w[:,c*] + h2o_b - L[c*] ----------------
__global__ __launch_bounds__(256) void out_kernel(const float* __restrict__ w3,
                                                  const float* __restrict__ b3,
                                                  const int* __restrict__ aidx,
                                                  const float* __restrict__ L,
                                                  float* __restrict__ out) {
  const int row = blockIdx.x;
  const int c = aidx[row];
  const float l = L[c];
  for (int o = threadIdx.x; o < OUT_O; o += 256)
    out[(size_t)row * OUT_O + o] = w3[(size_t)o * CONC + c] + b3[o] - l;
}

extern "C" void kernel_launch(void* const* d_in, const int* in_sizes, int n_in,
                              void* d_out, int out_size, void* d_ws, size_t ws_size,
                              hipStream_t stream) {
  (void)in_sizes; (void)n_in; (void)out_size; (void)ws_size;
  const float* x  = (const float*)d_in[0];
  // d_in[1] = dummy "hidden1"
  const float* w1 = (const float*)d_in[2];
  const float* b1 = (const float*)d_in[3];
  const float* w2 = (const float*)d_in[4];
  const float* b2 = (const float*)d_in[5];
  const float* w3 = (const float*)d_in[6];
  const float* b3 = (const float*)d_in[7];
  float* out = (float*)d_out;

  char* ws = (char*)d_ws;
  __bf16* w1b = (__bf16*)(ws);                            // 512*256*2   = 262144 B
  __bf16* w2b = (__bf16*)(ws + 262144);                   // 1024*512*2  = 1048576 B
  float*  Lt  = (float*)(ws + 262144 + 1048576);          // 1024*4      = 4096 B
  int*    ai  = (int*)(ws + 262144 + 1048576 + 4096);     // 32768*4     = 131072 B

  cvt_kernel<<<(HID * IN_I + 255) / 256, 256, 0, stream>>>(w1, w1b, HID * IN_I);
  cvt_kernel<<<(CONC * HID + 255) / 256, 256, 0, stream>>>(w2, w2b, CONC * HID);
  lse_kernel<<<CONC, 256, 0, stream>>>(w3, b3, Lt);

  // 4 waves/block * 16 rows = 64 rows/block; LDS = 66560 + 16896 + 33280 = 116736 B
  ff_wta_kernel<<<M_ROWS / 64, 128, 116736, stream>>>(x, w1b, b1, w2b, b2, ai);

  out_kernel<<<M_ROWS, 256, 0, stream>>>(w3, b3, ai, Lt, out);
}
